// MoEINT4_61881888801247
// MI455X (gfx1250) — matrix operand order repaired and checked by reading, not compiled, not run on hardware
//
#include <hip/hip_runtime.h>
#include <hip/hip_bf16.h>

typedef __attribute__((ext_vector_type(16))) __bf16 v16bf;
typedef __attribute__((ext_vector_type(8)))  __bf16 v8bf;
typedef __attribute__((ext_vector_type(8)))  float  v8f;

#define E_ 8
#define T_ 8192
#define H_ 2048
#define F_ 8192
#define P_ (H_ / 2)

#define MT  64            // token rows per block tile
#define NT  128           // feature cols per block tile
#define KT  64            // K per LDS stage (2 WMMA k-substeps)
#define LDK 72            // padded LDS row stride (bf16 elems) = 144 B, 16B-aligned rows
#define NIT (H_ / KT)     // 32 stages

// ---------------------------------------------------------------------------
// Pass 1: rowsum[t] = sum_h x[t,h]  (folds zero-point AND the +128 bias trick)
// ---------------------------------------------------------------------------
__global__ __launch_bounds__(256)
void MoEINT4_rowsum_kernel(const float* __restrict__ x, float* __restrict__ rs) {
    __shared__ float red[8];
    const int t = blockIdx.x;
    const float* row = x + (size_t)t * H_;
    float s = 0.f;
    for (int c = threadIdx.x; c < H_; c += 256) s += row[c];
    #pragma unroll
    for (int o = 16; o > 0; o >>= 1) s += __shfl_down(s, o, 32);
    if ((threadIdx.x & 31) == 0) red[threadIdx.x >> 5] = s;
    __syncthreads();
    if (threadIdx.x == 0) {
        float tot = 0.f;
        #pragma unroll
        for (int i = 0; i < 8; ++i) tot += red[i];
        rs[t] = tot;
    }
}

// ---------------------------------------------------------------------------
// Staging helpers
// ---------------------------------------------------------------------------
struct ATile { float4 f[4]; };   // 16 fp32
struct BTile { int4   p[4]; };   // 16 packed int32 (1 byte = 2 nibbles each)

__device__ __forceinline__ v8bf cvt8(const float4& f0, const float4& f1) {
    v8bf r;
    r[0] = (__bf16)f0.x; r[1] = (__bf16)f0.y; r[2] = (__bf16)f0.z; r[3] = (__bf16)f0.w;
    r[4] = (__bf16)f1.x; r[5] = (__bf16)f1.y; r[6] = (__bf16)f1.z; r[7] = (__bf16)f1.w;
    return r;
}
// bf16(128+n) == 0x4300 | n  (exact: 7-bit mantissa holds the nibble).
// One packed byte -> one dword {bf16(128+hi), bf16(128+lo)} with integer ops only.
__device__ __forceinline__ unsigned nibpair(unsigned v) {
    return 0x43004300u | (v & 0xFu) | ((v & 0xF0u) << 12);
}
__device__ __forceinline__ uint4 nib16(const int4& p) {
    uint4 r;
    r.x = nibpair((unsigned)p.x);
    r.y = nibpair((unsigned)p.y);
    r.z = nibpair((unsigned)p.z);
    r.w = nibpair((unsigned)p.w);
    return r;
}
__device__ __forceinline__ void loadA(ATile& t, const float* p, bool valid) {
    if (valid) {
        t.f[0] = ((const float4*)p)[0]; t.f[1] = ((const float4*)p)[1];
        t.f[2] = ((const float4*)p)[2]; t.f[3] = ((const float4*)p)[3];
    } else {
        t.f[0] = t.f[1] = t.f[2] = t.f[3] = make_float4(0.f, 0.f, 0.f, 0.f);
    }
}
__device__ __forceinline__ void loadB(BTile& t, const int* p) {
    t.p[0] = ((const int4*)p)[0]; t.p[1] = ((const int4*)p)[1];
    t.p[2] = ((const int4*)p)[2]; t.p[3] = ((const int4*)p)[3];
}
__device__ __forceinline__ void storeA(__bf16* dst, const ATile& t) {
    *(v8bf*)(dst)     = cvt8(t.f[0], t.f[1]);
    *(v8bf*)(dst + 8) = cvt8(t.f[2], t.f[3]);
}
__device__ __forceinline__ void storeB(__bf16* dst, const BTile& t) {
    uint4* d = (uint4*)dst;      // 16 int32 -> 32 bf16 (biased by +128)
    d[0] = nib16(t.p[0]);
    d[1] = nib16(t.p[1]);
    d[2] = nib16(t.p[2]);
    d[3] = nib16(t.p[3]);
}

// ---------------------------------------------------------------------------
// Pass 2: grouped int4 GEMM via bf16 WMMA, double-buffered LDS pipeline.
// acc = sum_h x*(q+128)  =>  out = scale * ( acc - (zp+128)*rowsum )
// ---------------------------------------------------------------------------
__global__ __launch_bounds__(256)
void MoEINT4_wmma_kernel(const int*   __restrict__ pw,
                         const float* __restrict__ scales,
                         const float* __restrict__ zps,
                         const float* __restrict__ x,
                         const int*   __restrict__ tpe,
                         const int*   __restrict__ ioff,
                         const float* __restrict__ rowsum,
                         float*       __restrict__ out) {
    const int e  = blockIdx.y >> 7;          // gridDim.y = E * (T/MT)
    const int rt = blockIdx.y & 127;
    const int ne = tpe[e];
    if (rt * MT >= ne) return;               // block-uniform early exit
    const int off       = ioff[e];
    const int rowsValid = min(MT, ne - rt * MT);
    const int tokBase   = off + rt * MT;
    const int fBase     = blockIdx.x * NT;

    __shared__ __attribute__((aligned(16))) __bf16 sA[2][MT][LDK];
    __shared__ __attribute__((aligned(16))) __bf16 sB[2][NT][LDK];
    __shared__ float smRS[MT];

    const int tid    = threadIdx.x;
    const int lane   = tid & 31;
    const int wv     = tid >> 5;
    const int mw     = wv & 1;               // 2 wave rows   (32 tokens each)
    const int nw     = wv >> 1;              // 4 wave cols   (32 features each)
    const int laneN  = lane & 15;
    const int laneHi = lane >> 4;

    v8f acc[2][2] = {};

    // staging assignment: A 64x64 fp32 (16/thread), B 128x32 int32 (16/thread)
    const int arow = tid >> 2, acol = (tid & 3) << 4;
    const int brow = tid >> 1, bcol = (tid & 1) << 4;
    const bool aValid = arow < rowsValid;
    const float* aPtr = x  + (size_t)(tokBase + arow) * H_ + acol;
    const int*   bPtr = pw + ((size_t)e * F_ + (size_t)(fBase + brow)) * P_ + bcol;

    ATile at; BTile bt;
    loadA(at, aPtr, aValid);
    loadB(bt, bPtr);
    storeA(&sA[0][arow][acol], at);
    storeB(&sB[0][brow][bcol * 2], bt);
    __syncthreads();

    #pragma unroll 2
    for (int it = 0; it < NIT; ++it) {
        const int cur = it & 1;
        const int nxt = cur ^ 1;
        const bool more = (it + 1 < NIT);

        // issue next tile's global loads before compute (latency hidden by WMMA)
        if (more) {
            loadA(at, aPtr + (it + 1) * KT, aValid);
            loadB(bt, bPtr + (it + 1) * (KT / 2));
        }

        // 2 k-substeps x (2x2) WMMA tiles = 8 WMMAs per wave per stage
        #pragma unroll
        for (int ks = 0; ks < 2; ++ks) {
            v16bf afr[2], bfr[2];
            const int aOff = ks * 32 + (laneHi ? 8 : 0);
            const int bOff = ks * 32 + laneHi * 8;
            #pragma unroll
            for (int i = 0; i < 2; ++i) {
                const __bf16* ap = &sA[cur][mw * 32 + i * 16 + laneN][aOff];
                const v8bf alo = *(const v8bf*)ap;
                const v8bf ahi = *(const v8bf*)(ap + 16);
                #pragma unroll
                for (int q = 0; q < 8; ++q) { afr[i][q] = alo[q]; afr[i][q + 8] = ahi[q]; }
            }
            #pragma unroll
            for (int j = 0; j < 2; ++j) {
                const __bf16* bp = &sB[cur][nw * 32 + j * 16 + laneN][bOff];
                const v8bf blo = *(const v8bf*)bp;
                const v8bf bhi = *(const v8bf*)(bp + 16);
                #pragma unroll
                for (int q = 0; q < 8; ++q) { bfr[j][q] = blo[q]; bfr[j][q + 8] = bhi[q]; }
            }
            #pragma unroll
            for (int i = 0; i < 2; ++i)
                #pragma unroll
                for (int j = 0; j < 2; ++j)
                    acc[i][j] = __builtin_amdgcn_wmma_f32_16x16x32_bf16(
                        false, afr[i], false, bfr[j], (short)0, acc[i][j], false, false);
        }

        // convert + store next tile into the other buffer
        if (more) {
            storeA(&sA[nxt][arow][acol], at);
            storeB(&sB[nxt][brow][bcol * 2], bt);
        }
        __syncthreads();
    }

    // ---- epilogue ----
    if (tid < MT) smRS[tid] = (tid < rowsValid) ? rowsum[tokBase + tid] : 0.f;
    __syncthreads();

    float scv[2], zpv[2];
    #pragma unroll
    for (int j = 0; j < 2; ++j) {
        const int ncol = fBase + nw * 32 + j * 16 + laneN;
        scv[j] = scales[e * F_ + ncol];
        zpv[j] = zps[e * F_ + ncol] + 128.f;   // undo the +128 mantissa-bias trick
    }

    if (rowsValid == MT) {                    // fast path: no per-element masking
        #pragma unroll
        for (int i = 0; i < 2; ++i) {
            const int mBase = mw * 32 + i * 16 + laneHi * 8;
            #pragma unroll
            for (int r = 0; r < 8; ++r) {
                const int m = mBase + r;
                const float rsv = smRS[m];
                float* orow = out + (size_t)(tokBase + m) * F_;
                #pragma unroll
                for (int j = 0; j < 2; ++j) {
                    const int ncol = fBase + nw * 32 + j * 16 + laneN;
                    orow[ncol] = scv[j] * (acc[i][j][r] - zpv[j] * rsv);
                }
            }
        }
    } else {                                  // boundary tile
        #pragma unroll
        for (int i = 0; i < 2; ++i) {
            const int mBase = mw * 32 + i * 16 + laneHi * 8;
            #pragma unroll
            for (int r = 0; r < 8; ++r) {
                const int m = mBase + r;
                if (m < rowsValid) {
                    const float rsv = smRS[m];
                    float* orow = out + (size_t)(tokBase + m) * F_;
                    #pragma unroll
                    for (int j = 0; j < 2; ++j) {
                        const int ncol = fBase + nw * 32 + j * 16 + laneN;
                        orow[ncol] = scv[j] * (acc[i][j][r] - zpv[j] * rsv);
                    }
                }
            }
        }
    }
}

// ---------------------------------------------------------------------------
extern "C" void kernel_launch(void* const* d_in, const int* in_sizes, int n_in,
                              void* d_out, int out_size, void* d_ws, size_t ws_size,
                              hipStream_t stream) {
    const int*   pw     = (const int*)d_in[0];
    const float* scales = (const float*)d_in[1];
    const float* zps    = (const float*)d_in[2];
    const float* x      = (const float*)d_in[3];
    const int*   tpe    = (const int*)d_in[5];
    const int*   ioff   = (const int*)d_in[6];
    float* out = (float*)d_out;
    float* rs  = (float*)d_ws;               // rowsum[T], 32 KB

    MoEINT4_rowsum_kernel<<<T_, 256, 0, stream>>>(x, rs);

    dim3 grid(F_ / NT, E_ * (T_ / MT));      // (64, 1024); empty tiles exit early
    MoEINT4_wmma_kernel<<<grid, 256, 0, stream>>>(pw, scales, zps, x, tpe, ioff, rs, out);
}
